// Generator_31490700214663
// MI455X (gfx1250) — matrix operand order repaired and checked by reading, not compiled, not run on hardware
//
#include <hip/hip_runtime.h>
#include <math.h>

typedef __attribute__((ext_vector_type(16))) _Float16 v16h;
typedef __attribute__((ext_vector_type(8)))  _Float16 v8h;
typedef __attribute__((ext_vector_type(8)))  float    v8f;

#define NB 32
#define LATENT 512

static __device__ __forceinline__ float leakyf(float v) { return v >= 0.f ? v : 0.2f * v; }

// ---------------- weight f32 -> f16 copy ----------------
__global__ __launch_bounds__(256) void k_f32_to_f16(const float* __restrict__ src,
                                                    _Float16* __restrict__ dst, int n) {
  int i = blockIdx.x * 256 + threadIdx.x;
  if (i < n) dst[i] = (_Float16)src[i];
}

// ---------------- mapping network: normalize + 4x (GEMM 32x512x512 + bias + leaky) ----------------
// Single block, 8 waves. Activations ping-pong in LDS as f16.
__global__ __launch_bounds__(256) void k_mapping(const float* __restrict__ z,
                                                 const _Float16* __restrict__ mWh, // [4][512 out][512 in]
                                                 const float* __restrict__ mapb,   // [4][512]
                                                 float* __restrict__ w_out)        // [32][512] f32
{
  __shared__ _Float16 actA[NB][LATENT];
  __shared__ _Float16 actB[NB][LATENT];
  int tid = threadIdx.x;
  // per-row 1/(||z||+1e-8), staged temporarily in actB space
  float* rnp = (float*)&actB[0][0];
  if (tid < NB) {
    float s = 0.f;
    const float* zr = z + tid * LATENT;
    for (int k = 0; k < LATENT; k++) { float v = zr[k]; s += v * v; }
    rnp[tid] = 1.0f / (sqrtf(s) + 1e-8f);
  }
  __syncthreads();
  for (int idx = tid; idx < NB * LATENT; idx += 256) {
    int b = idx >> 9, k = idx & 511;
    actA[b][k] = (_Float16)(z[idx] * rnp[b]);
  }
  __syncthreads();

  int wave = tid >> 5, lane = tid & 31, l = lane & 15, hi = lane >> 4;
  for (int layer = 0; layer < 4; layer++) {
    const _Float16* Wl = mWh + (size_t)layer * LATENT * LATENT; // [n][k]
    const float* bl = mapb + layer * LATENT;
    _Float16 (*src)[LATENT] = (layer & 1) ? actB : actA;
    _Float16 (*dst)[LATENT] = (layer & 1) ? actA : actB;
    for (int t = wave; t < 64; t += 8) {           // 2 M-tiles x 32 N-tiles
      int m0 = (t & 1) * 16;
      int n0 = (t >> 1) * 16;
      v8f c = {};
      const _Float16* arow = &src[m0 + l][0];
      const _Float16* brow = Wl + (size_t)(n0 + l) * LATENT;
      for (int ks = 0; ks < LATENT; ks += 32) {
        v8h a0 = *(const v8h*)(arow + ks + hi * 16);
        v8h a1 = *(const v8h*)(arow + ks + hi * 16 + 8);
        v16h a;
#pragma unroll
        for (int q = 0; q < 8; q++) { a[q] = a0[q]; a[q + 8] = a1[q]; }
        v16h bf = *(const v16h*)(brow + ks + hi * 16); // B[k][n] = W[n][k], contiguous in k
        c = __builtin_amdgcn_wmma_f32_16x16x32_f16(false, a, false, bf, (short)0, c, false, false);
      }
      int n = n0 + l;
      float bias = bl[n];
#pragma unroll
      for (int g = 0; g < 8; g++) {
        int m = m0 + hi * 8 + g;
        dst[m][n] = (_Float16)leakyf(c[g] + bias);
      }
    }
    __syncthreads();
  }
  // final result is in actA (layer 3 wrote actA)
  for (int idx = tid; idx < NB * LATENT; idx += 256) {
    int b = idx >> 9, k = idx & 511;
    w_out[idx] = (float)actA[b][k];
  }
}

// ---------------- style modulation factors: modf[b][i] = 1 + w@sW.T + sb ----------------
__global__ __launch_bounds__(256) void k_styles(const float* __restrict__ w,
                                                const float* __restrict__ sW,
                                                const float* __restrict__ sb,
                                                float* __restrict__ modf, int cin) {
  int i = blockIdx.x * 256 + threadIdx.x;
  if (i >= NB * cin) return;
  int b = i / cin, row = i - b * cin;
  const float* wr = w + b * LATENT;
  const float* sr = sW + (size_t)row * LATENT;
  float acc = 0.f;
  for (int k = 0; k < LATENT; k++) acc += wr[k] * sr[k];
  modf[i] = 1.0f + acc + sb[row];
}

// ---------------- q[o][i] = sum_k cw[o][i][k]^2 ----------------
__global__ __launch_bounds__(256) void k_q(const float* __restrict__ cw, float* __restrict__ q, int n) {
  int i = blockIdx.x * 256 + threadIdx.x;
  if (i >= n) return;
  const float* p = cw + (size_t)i * 9;
  float s = 0.f;
#pragma unroll
  for (int r = 0; r < 9; r++) s += p[r] * p[r];
  q[i] = s;
}

// ---------------- demod[b][o] = rsqrt( (1/cin) * sum_i modf^2 * q + 1e-8 ) ----------------
__global__ __launch_bounds__(256) void k_demod(const float* __restrict__ modf,
                                               const float* __restrict__ q,
                                               float* __restrict__ dem, int cout, int cin) {
  int i = blockIdx.x * 256 + threadIdx.x;
  if (i >= NB * cout) return;
  int b = i / cout, o = i - b * cout;
  const float* mf = modf + b * cin;
  const float* qo = q + (size_t)o * cin;
  float s = 0.f;
  for (int c = 0; c < cin; c++) { float m = mf[c]; s += m * m * qo[c]; }
  dem[i] = rsqrtf(s / (float)cin + 1e-8f);
}

// ---------------- pack conv weights: wp[o][r*cin + i] = cw[o][i][r] * scale (f16) ----------------
__global__ __launch_bounds__(256) void k_wpack(const float* __restrict__ cw, _Float16* __restrict__ wp,
                                               int cout, int cin, float scale) {
  int total = cout * 9 * cin;
  int i = blockIdx.x * 256 + threadIdx.x;
  if (i >= total) return;
  int o = i / (9 * cin);
  int rem = i - o * 9 * cin;
  int r = rem / cin;
  int ci = rem - r * cin;
  wp[i] = (_Float16)(cw[((size_t)o * cin + ci) * 9 + r] * scale);
}

// ---------------- layer 0 modulated input from const (4x4) ----------------
__global__ __launch_bounds__(256) void k_mod0(const float* __restrict__ cst,
                                              const float* __restrict__ modf,
                                              _Float16* __restrict__ xm) {
  int i = blockIdx.x * 256 + threadIdx.x;
  if (i >= NB * 512 * 16) return;
  int b = i / (512 * 16);
  int rem = i - b * 512 * 16;
  int c = rem >> 4, p = rem & 15;
  xm[i] = (_Float16)(cst[c * 16 + p] * modf[b * 512 + c]);
}

// ---------------- bilinear 2x upsample (half-pixel) + modulate -> f16 ----------------
__global__ __launch_bounds__(256) void k_mod_up(const float* __restrict__ xin,
                                                const float* __restrict__ modf,
                                                _Float16* __restrict__ xm, int cin, int Hi) {
  int Ho = Hi * 2;
  int total = NB * cin * Ho * Ho;
  int i = blockIdx.x * 256 + threadIdx.x;
  if (i >= total) return;
  int xo = i % Ho; int t = i / Ho;
  int yo = t % Ho; t /= Ho;
  int c = t % cin; int b = t / cin;
  int jy = yo >> 1, jx = xo >> 1;
  int ya = jy, yb = (yo & 1) ? (jy + 1 < Hi ? jy + 1 : Hi - 1) : (jy > 0 ? jy - 1 : 0);
  int xa = jx, xb = (xo & 1) ? (jx + 1 < Hi ? jx + 1 : Hi - 1) : (jx > 0 ? jx - 1 : 0);
  const float* p = xin + ((size_t)b * cin + c) * Hi * Hi;
  float v = 0.5625f * p[ya * Hi + xa] + 0.1875f * p[ya * Hi + xb] +
            0.1875f * p[yb * Hi + xa] + 0.0625f * p[yb * Hi + xb];
  xm[i] = (_Float16)(v * modf[b * cin + c]);
}

// ---------------- implicit-GEMM modulated conv via WMMA f16 ----------------
// Each wave owns MB M-tiles (output-channel tiles) for ONE N-tile (pixel tile):
// one im2col B-fragment gather feeds MB WMMAs. Templated dims -> immediate
// load offsets and shift-based index math.
template <int TCIN, int TH, int TW, int MB>
__global__ __launch_bounds__(256) void k_conv(const _Float16* __restrict__ wp,   // [M][9*TCIN]
                                              const _Float16* __restrict__ xm,   // [NB][TCIN][TH][TW]
                                              const float* __restrict__ dem,     // [NB][M]
                                              const float* __restrict__ nw,      // [M]
                                              const float* __restrict__ noise,   // [NB][M][TH][TW]
                                              float* __restrict__ xout,          // [NB][M][TH][TW]
                                              int M) {
  constexpr int K = 9 * TCIN;
  constexpr int HW = TH * TW;
  const int mTiles = M >> 4;
  const int nTiles = (NB * HW) >> 4;
  const int mGroups = mTiles / MB;
  int wave = threadIdx.x >> 5, lane = threadIdx.x & 31, l = lane & 15, hi = lane >> 4;
  int tile = blockIdx.x * 8 + wave;
  if (tile >= mGroups * nTiles) return;          // wave-uniform: EXEC stays all-ones for WMMA
  int mg = tile % mGroups, nt = tile / mGroups;
  int n = nt * 16 + l;
  int b = n / HW;                                 // power-of-two -> shift
  int rem = n - b * HW;
  int y = rem / TW;
  int x = rem - y * TW;
  const _Float16* xmb = xm + (size_t)b * TCIN * HW;
  const _Float16* arow[MB];
#pragma unroll
  for (int mi = 0; mi < MB; mi++)
    arow[mi] = wp + (size_t)((mg * MB + mi) * 16 + l) * K;
  v8f c[MB] = {};
  for (int ks = 0; ks < K; ks += 32) {
    int kb = ks + hi * 16;
    int r = kb / TCIN;                            // tap 0..8, constant across this lane's 16 k's
    int i0 = kb % TCIN;
    int ky = (r >= 6) ? 2 : (r >= 3 ? 1 : 0);
    int kx = r - 3 * ky;
    int yy = y + ky - 1, xx = x + kx - 1;
    int inb = ((unsigned)yy < (unsigned)TH) & ((unsigned)xx < (unsigned)TW);
    int yc = inb ? yy : 0, xc = inb ? xx : 0;
    _Float16 msk = (_Float16)(float)inb;          // branchless zero-padding
    const _Float16* p = xmb + (size_t)i0 * HW + yc * TW + xc;
    v16h bf;
#pragma unroll
    for (int q = 0; q < 16; q++) bf[q] = p[q * HW] * msk;   // immediate-offset d16 clause
    __builtin_prefetch(p + 16 * HW, 0, 3);        // global_prefetch_b8: next k-step's gather base
#pragma unroll
    for (int mi = 0; mi < MB; mi++) {
      v8h a0 = *(const v8h*)(arow[mi] + ks + hi * 16);
      v8h a1 = *(const v8h*)(arow[mi] + ks + hi * 16 + 8);
      v16h a;
#pragma unroll
      for (int q = 0; q < 8; q++) { a[q] = a0[q]; a[q + 8] = a1[q]; }
      c[mi] = __builtin_amdgcn_wmma_f32_16x16x32_f16(false, a, false, bf, (short)0, c[mi], false, false);
    }
  }
  const float* demb = dem + b * M;
  const float* noib = noise + (size_t)b * M * HW + y * TW + x;
  float* outb = xout + (size_t)b * M * HW + y * TW + x;
#pragma unroll
  for (int mi = 0; mi < MB; mi++) {
    int obase = (mg * MB + mi) * 16 + hi * 8;
#pragma unroll
    for (int g = 0; g < 8; g++) {
      int o = obase + g;
      float v = c[mi][g] * demb[o] + nw[o] * noib[(size_t)o * HW];
      outb[(size_t)o * HW] = leakyf(v);
    }
  }
}

// ---------------- ToRGB 1x1 conv + tanh ----------------
__global__ __launch_bounds__(256) void k_torgb(const float* __restrict__ xin,  // [NB][32][64][64]
                                               const float* __restrict__ rgbW, // [3][32]
                                               const float* __restrict__ rgbB, // [3]
                                               float* __restrict__ out) {      // [NB][3][64][64]
  int i = blockIdx.x * 256 + threadIdx.x;
  const int P = 64 * 64;
  if (i >= NB * 3 * P) return;
  int b = i / (3 * P);
  int rem = i - b * 3 * P;
  int o = rem / P;
  int p = rem - o * P;
  const float* xb = xin + (size_t)b * 32 * P + p;
  float s = rgbB[o];
#pragma unroll
  for (int ci = 0; ci < 32; ci++) s += xb[(size_t)ci * P] * rgbW[o * 32 + ci];
  out[i] = tanhf(s);
}

extern "C" void kernel_launch(void* const* d_in, const int* in_sizes, int n_in,
                              void* d_out, int out_size, void* d_ws, size_t ws_size,
                              hipStream_t stream) {
  (void)in_sizes; (void)n_in; (void)out_size; (void)ws_size;
  const float* z    = (const float*)d_in[0];
  const float* mapW = (const float*)d_in[1];
  const float* mapb = (const float*)d_in[2];
  const float* cst  = (const float*)d_in[3];
  const float *styW[5], *styB[5], *convW[5], *nseW[5], *nse[5];
  for (int i = 0; i < 5; i++) {
    styW[i]  = (const float*)d_in[4 + i];
    styB[i]  = (const float*)d_in[9 + i];
    convW[i] = (const float*)d_in[14 + i];
    nseW[i]  = (const float*)d_in[19 + i];
    nse[i]   = (const float*)d_in[24 + i];
  }
  const float* rgbW = (const float*)d_in[29];
  const float* rgbB = (const float*)d_in[30];
  float* out = (float*)d_out;

  static const int CIN[5]  = {512, 512, 256, 128, 64};
  static const int COUT[5] = {512, 256, 128, 64, 32};
  static const int cumCin[5]  = {0, 512, 1024, 1280, 1408};           // total 1472
  static const int cumQ[5]    = {0, 262144, 393216, 425984, 434176};  // total 436224
  static const int cumCout[5] = {0, 512, 768, 896, 960};              // total 992

  char* ws = (char*)d_ws;
  size_t off = 0;
  auto carve = [&](size_t bytes) -> char* {
    char* p = ws + off;
    off = (off + bytes + 255) & ~(size_t)255;
    return p;
  };
  float*    w_f  = (float*)carve((size_t)NB * 512 * 4);
  _Float16* mWh  = (_Float16*)carve((size_t)4 * 512 * 512 * 2);
  float*    modf = (float*)carve((size_t)NB * 1472 * 4);
  float*    qb   = (float*)carve((size_t)436224 * 4);
  float*    dem  = (float*)carve((size_t)NB * 992 * 4);
  _Float16* wpk  = (_Float16*)carve((size_t)436224 * 9 * 2);
  _Float16* xmb  = (_Float16*)carve((size_t)NB * 64 * 64 * 64 * 2);
  float*    xA   = (float*)carve((size_t)NB * 32 * 64 * 64 * 4);
  float*    xB   = (float*)carve((size_t)NB * 32 * 64 * 64 * 4);

  // mapping network
  k_f32_to_f16<<<(4 * 512 * 512 + 255) / 256, 256, 0, stream>>>(mapW, mWh, 4 * 512 * 512);
  k_mapping<<<1, 256, 0, stream>>>(z, mWh, mapb, w_f);

  // per-layer style/demod/weight prep
  for (int i = 0; i < 5; i++) {
    int cin = CIN[i], cout = COUT[i];
    float scale = 1.0f / sqrtf((float)cin);
    k_styles<<<(NB * cin + 255) / 256, 256, 0, stream>>>(w_f, styW[i], styB[i], modf + NB * cumCin[i], cin);
    k_q<<<(cout * cin + 255) / 256, 256, 0, stream>>>(convW[i], qb + cumQ[i], cout * cin);
    k_demod<<<(NB * cout + 255) / 256, 256, 0, stream>>>(modf + NB * cumCin[i], qb + cumQ[i],
                                                         dem + NB * cumCout[i], cout, cin);
    k_wpack<<<(cout * 9 * cin + 255) / 256, 256, 0, stream>>>(convW[i], wpk + (size_t)cumQ[i] * 9,
                                                              cout, cin, scale);
  }

  // block 0: const 4x4, M=512 (8 m-groups of 4) x 32 n-tiles = 256 tiles -> 32 blocks
  k_mod0<<<(NB * 512 * 16 + 255) / 256, 256, 0, stream>>>(cst, modf, xmb);
  k_conv<512, 4, 4, 4><<<32, 256, 0, stream>>>(wpk, xmb, dem, nseW[0], nse[0], xA, 512);

  // block 1: 8x8, M=256: 4 m-groups x 128 n-tiles = 512 tiles -> 64 blocks
  k_mod_up<<<((size_t)NB * 512 * 64 + 255) / 256, 256, 0, stream>>>(xA, modf + NB * cumCin[1], xmb, 512, 4);
  k_conv<512, 8, 8, 4><<<64, 256, 0, stream>>>(wpk + (size_t)cumQ[1] * 9, xmb, dem + NB * cumCout[1],
                                               nseW[1], nse[1], xB, 256);

  // block 2: 16x16, M=128: 2 m-groups x 512 n-tiles = 1024 tiles -> 128 blocks
  k_mod_up<<<((size_t)NB * 256 * 256 + 255) / 256, 256, 0, stream>>>(xB, modf + NB * cumCin[2], xmb, 256, 8);
  k_conv<256, 16, 16, 4><<<128, 256, 0, stream>>>(wpk + (size_t)cumQ[2] * 9, xmb, dem + NB * cumCout[2],
                                                  nseW[2], nse[2], xA, 128);

  // block 3: 32x32, M=64: 1 m-group x 2048 n-tiles = 2048 tiles -> 256 blocks
  k_mod_up<<<((size_t)NB * 128 * 1024 + 255) / 256, 256, 0, stream>>>(xA, modf + NB * cumCin[3], xmb, 128, 16);
  k_conv<128, 32, 32, 4><<<256, 256, 0, stream>>>(wpk + (size_t)cumQ[3] * 9, xmb, dem + NB * cumCout[3],
                                                  nseW[3], nse[3], xB, 64);

  // block 4: 64x64, M=32 (MB=2): 1 m-group x 8192 n-tiles = 8192 tiles -> 1024 blocks
  k_mod_up<<<((size_t)NB * 64 * 4096 + 255) / 256, 256, 0, stream>>>(xB, modf + NB * cumCin[4], xmb, 64, 32);
  k_conv<64, 64, 64, 2><<<1024, 256, 0, stream>>>(wpk + (size_t)cumQ[4] * 9, xmb, dem + NB * cumCout[4],
                                                  nseW[4], nse[4], xA, 32);

  // ToRGB + tanh
  k_torgb<<<(NB * 3 * 64 * 64 + 255) / 256, 256, 0, stream>>>(xA, rgbW, rgbB, out);
}
